// LiquidNeuralNetwork_27307402068100
// MI455X (gfx1250) — matrix operand order repaired and checked by reading, not compiled, not run on hardware
//
#include <hip/hip_runtime.h>

// Problem sizes (fixed by the reference).
#define Bsz 64
#define Ssz 512
#define Isz 512
#define Hsz 1024
#define Osz 512
#define KSPLIT 4   // K-split factor for the latency-bound scan GEMMs

typedef __attribute__((ext_vector_type(16))) _Float16 v16h;
typedef __attribute__((ext_vector_type(8)))  _Float16 v8h;
typedef __attribute__((ext_vector_type(8)))  float    v8f;

// ---------------------------------------------------------------------------
// f32 -> f16 conversion (one-time for x and all weights)
// ---------------------------------------------------------------------------
__global__ void cvt_f32_to_f16(const float* __restrict__ src,
                               _Float16* __restrict__ dst, long long n) {
  long long i = (long long)blockIdx.x * blockDim.x + threadIdx.x;
  long long stride = (long long)gridDim.x * blockDim.x;
  for (; i < n; i += stride) dst[i] = (_Float16)src[i];
}

__global__ void zero_state(float* __restrict__ h, _Float16* __restrict__ h16, int n) {
  int i = blockIdx.x * blockDim.x + threadIdx.x;
  if (i < n) { h[i] = 0.0f; h16[i] = (_Float16)0.0f; }
}

__global__ void copy_f32(const float* __restrict__ src, float* __restrict__ dst, int n) {
  int i = blockIdx.x * blockDim.x + threadIdx.x;
  if (i < n) dst[i] = src[i];
}

// ---------------------------------------------------------------------------
// Register-blocked WMMA GEMM:
//   C[M,N] = A[M,K](f16,row-major,lda) @ W[N,K]^T(f16,row-major,ldw) [+ bias]
// Each wave computes an MB x NB block of 16x16 tiles: MB*NB WMMAs per
// 2*(MB+NB) b128 loads, MB*NB independent accumulator chains.
// PARTIAL=true : K-split slice (gridDim.y slices); raw f32 partial written to
//                Cpart[kz][M*N], no bias (straight-line epilogue).
// PARTIAL=false: gridDim.y==1; bias added; f16 result written to Ch (ldc).
// Fragment layouts per CDNA5 ISA 7.12.2:
//   A 16x32 f16: lane L (row=L&15, grp=L>>4): e[j]=A[row][(j<8?0:16)+grp*8+(j&7)]
//   B 32x16 f16: lane L (col=L&15): e[j]=W[col][(j<8?0:16)+grp*8+(j&7)]
//   C/D f32   : c[e] -> row m0+e+grp*8, col n0+(L&15)
// ---------------------------------------------------------------------------
template <int MB, int NB, bool PARTIAL>
__global__ void wmma_gemm_f16(const _Float16* __restrict__ A, long long lda,
                              const _Float16* __restrict__ W, long long ldw,
                              const float* __restrict__ bias,
                              _Float16* __restrict__ Ch, long long ldc,
                              int M, int N, int K,
                              float* __restrict__ Cpart) {
  const int wavesPerBlock = blockDim.x >> 5;
  const int wave = threadIdx.x >> 5;
  const int lane = threadIdx.x & 31;
  const int grp  = lane >> 4;   // half-wave group
  const int r    = lane & 15;   // row (A) / col (B,C) within tile
  const int NG = (N >> 4) / NB;
  const int groups = ((M >> 4) / MB) * NG;
  const int kz = blockIdx.y;
  const int Kc = K / gridDim.y;        // K per split (divides evenly here)
  const int kbeg = kz * Kc;
  const v8f vzero = {0.f, 0.f, 0.f, 0.f, 0.f, 0.f, 0.f, 0.f};

  for (int g = blockIdx.x * wavesPerBlock + wave; g < groups;
       g += gridDim.x * wavesPerBlock) {
    const int mg = g / NG;
    const int ng = g - mg * NG;
    const int m0 = mg * MB * 16;
    const int n0 = ng * NB * 16;

    const _Float16* arow[MB];
    const _Float16* wrow[NB];
#pragma unroll
    for (int i = 0; i < MB; ++i)
      arow[i] = A + (size_t)(m0 + i * 16 + r) * lda + kbeg;
#pragma unroll
    for (int j = 0; j < NB; ++j)
      wrow[j] = W + (size_t)(n0 + j * 16 + r) * ldw + kbeg;

    v8f acc[MB][NB];
#pragma unroll
    for (int i = 0; i < MB; ++i)
#pragma unroll
      for (int j = 0; j < NB; ++j) acc[i][j] = vzero;

#pragma unroll 2
    for (int k0 = 0; k0 < Kc; k0 += 32) {
      v16h a[MB], b[NB];
#pragma unroll
      for (int i = 0; i < MB; ++i) {
        v8h lo = *(const v8h*)(arow[i] + k0 + grp * 8);
        v8h hi = *(const v8h*)(arow[i] + k0 + 16 + grp * 8);
        a[i] = __builtin_shufflevector(lo, hi, 0, 1, 2, 3, 4, 5, 6, 7,
                                       8, 9, 10, 11, 12, 13, 14, 15);
      }
#pragma unroll
      for (int j = 0; j < NB; ++j) {
        v8h lo = *(const v8h*)(wrow[j] + k0 + grp * 8);
        v8h hi = *(const v8h*)(wrow[j] + k0 + 16 + grp * 8);
        b[j] = __builtin_shufflevector(lo, hi, 0, 1, 2, 3, 4, 5, 6, 7,
                                       8, 9, 10, 11, 12, 13, 14, 15);
      }
#pragma unroll
      for (int i = 0; i < MB; ++i)
#pragma unroll
        for (int j = 0; j < NB; ++j)
          acc[i][j] = __builtin_amdgcn_wmma_f32_16x16x32_f16(
              false, a[i], false, b[j], (short)0, acc[i][j], false, false);
    }

    if (PARTIAL) {
      float* dst = Cpart + (size_t)kz * M * N;
#pragma unroll
      for (int i = 0; i < MB; ++i)
#pragma unroll
        for (int j = 0; j < NB; ++j) {
          const int col = n0 + j * 16 + r;
#pragma unroll
          for (int e = 0; e < 8; ++e) {
            const int row = m0 + i * 16 + e + grp * 8;
            dst[(size_t)row * N + col] = acc[i][j][e];
          }
        }
    } else {
#pragma unroll
      for (int i = 0; i < MB; ++i)
#pragma unroll
        for (int j = 0; j < NB; ++j) {
          const int col = n0 + j * 16 + r;
          const float bv = bias[col];
#pragma unroll
          for (int e = 0; e < 8; ++e) {
            const int row = m0 + i * 16 + e + grp * 8;
            Ch[(size_t)row * ldc + col] = (_Float16)(acc[i][j][e] + bv);
          }
        }
    }
  }
}

// ---------------------------------------------------------------------------
// Fused GRU gates + liquid update + LayerNorm. One block per batch row.
// Sums the KSPLIT gh partials (+ b_hh) inline, then:
//   r = sig(xr+hr); z = sig(xz+hz); n = tanh(xn + r*hn)
//   h_new = (1-z)*n + z*h;  h = h + (h_new - h)/tau;  h = LN(h)*gamma + beta
// Writes h (f32) and h16 (f16 copy for the next WMMA GEMMs).
// ---------------------------------------------------------------------------
__global__ void gru_gate_ln(const float* __restrict__ ghp,   // [KSPLIT][B*3H]
                            const float* __restrict__ b_hh,
                            const _Float16* __restrict__ gx_t, long long gx_rs,
                            const float* __restrict__ tau,
                            const float* __restrict__ gamma,
                            const float* __restrict__ beta,
                            float* __restrict__ h, _Float16* __restrict__ h16) {
  const int b = blockIdx.x;
  const int tid = threadIdx.x;
  constexpr int PER = Hsz / 256;

  const _Float16* gxr = gx_t + (long long)b * gx_rs;
  float* hrow = h + (long long)b * Hsz;
  _Float16* h16r = h16 + (long long)b * Hsz;

  float vals[PER];
  float s = 0.f, sq = 0.f;
#pragma unroll
  for (int c = 0; c < PER; ++c) {
    const int j = tid + c * 256;
    float hr = b_hh[j], hz = b_hh[Hsz + j], hn = b_hh[2 * Hsz + j];
#pragma unroll
    for (int k = 0; k < KSPLIT; ++k) {
      const float* p = ghp + (size_t)k * (Bsz * 3 * Hsz) + (size_t)b * 3 * Hsz;
      hr += p[j];
      hz += p[Hsz + j];
      hn += p[2 * Hsz + j];
    }
    const float xr = (float)gxr[j];
    const float xz = (float)gxr[Hsz + j];
    const float xn = (float)gxr[2 * Hsz + j];
    const float rg = 1.0f / (1.0f + __expf(-(xr + hr)));
    const float zg = 1.0f / (1.0f + __expf(-(xz + hz)));
    const float ng = tanhf(xn + rg * hn);
    const float hp = hrow[j];
    const float hnew = (1.0f - zg) * ng + zg * hp;
    const float hl = hp + (hnew - hp) / tau[j];
    vals[c] = hl;
    s += hl;
    sq += hl * hl;
  }

  __shared__ float s1[256], s2[256];
  s1[tid] = s; s2[tid] = sq;
  __syncthreads();
  for (int off = 128; off > 0; off >>= 1) {
    if (tid < off) { s1[tid] += s1[tid + off]; s2[tid] += s2[tid + off]; }
    __syncthreads();
  }
  const float mu = s1[0] * (1.0f / Hsz);
  const float var = s2[0] * (1.0f / Hsz) - mu * mu;
  const float rs = rsqrtf(var + 1e-5f);

#pragma unroll
  for (int c = 0; c < PER; ++c) {
    const int j = tid + c * 256;
    const float y = (vals[c] - mu) * rs * gamma[j] + beta[j];
    hrow[j] = y;
    h16r[j] = (_Float16)y;
  }
}

// ---------------------------------------------------------------------------
// Reduce KSPLIT out-partials (+ bias) into d_out at [b, t, :].
// dst points at d_out + t*O; element (b,o) lands at dst[b*S*O + o].
// ---------------------------------------------------------------------------
__global__ void out_reduce(const float* __restrict__ part,   // [KSPLIT][B*O]
                           const float* __restrict__ bias,
                           float* __restrict__ dst) {
  const int i = blockIdx.x * blockDim.x + threadIdx.x;
  if (i >= Bsz * Osz) return;
  const int b = i / Osz;
  const int o = i - b * Osz;
  float s = bias[o];
#pragma unroll
  for (int k = 0; k < KSPLIT; ++k) s += part[(size_t)k * (Bsz * Osz) + i];
  dst[(size_t)b * Ssz * Osz + o] = s;
}

// ---------------------------------------------------------------------------
extern "C" void kernel_launch(void* const* d_in, const int* in_sizes, int n_in,
                              void* d_out, int out_size, void* d_ws, size_t ws_size,
                              hipStream_t stream) {
  (void)in_sizes; (void)n_in; (void)out_size; (void)ws_size;

  const float* x     = (const float*)d_in[0];
  const float* W_in  = (const float*)d_in[1];
  const float* b_in  = (const float*)d_in[2];
  const float* W_ih  = (const float*)d_in[3];
  const float* b_ih  = (const float*)d_in[4];
  const float* W_hh  = (const float*)d_in[5];
  const float* b_hh  = (const float*)d_in[6];
  const float* W_out = (const float*)d_in[7];
  const float* b_out = (const float*)d_in[8];
  const float* gamma = (const float*)d_in[9];
  const float* beta  = (const float*)d_in[10];
  const float* tau   = (const float*)d_in[11];
  float* out = (float*)d_out;

  // Workspace carve-up (~306 MB total).
  char* ws = (char*)d_ws;
  size_t off = 0;
  auto carve = [&](size_t bytes) -> char* {
    char* p = ws + off;
    off += (bytes + 255) & ~(size_t)255;
    return p;
  };
  const long long BS = (long long)Bsz * Ssz;
  _Float16* xh    = (_Float16*)carve((size_t)BS * Isz * 2);              //  32 MB
  _Float16* Winh  = (_Float16*)carve((size_t)Hsz * Isz * 2);             //   1 MB
  _Float16* Wihh  = (_Float16*)carve((size_t)3 * Hsz * Hsz * 2);         //   6 MB
  _Float16* Whhh  = (_Float16*)carve((size_t)3 * Hsz * Hsz * 2);         //   6 MB
  _Float16* Wouth = (_Float16*)carve((size_t)Osz * Hsz * 2);             //   1 MB
  _Float16* inph  = (_Float16*)carve((size_t)BS * Hsz * 2);              //  64 MB
  _Float16* gxh   = (_Float16*)carve((size_t)BS * 3 * Hsz * 2);          // 192 MB
  float*    hbuf  = (float*)carve((size_t)Bsz * Hsz * 4);
  _Float16* h16   = (_Float16*)carve((size_t)Bsz * Hsz * 2);
  float*    ghp   = (float*)carve((size_t)KSPLIT * Bsz * 3 * Hsz * 4);   //   3 MB
  float*    outp  = (float*)carve((size_t)KSPLIT * Bsz * Osz * 4);       // 0.5 MB

  auto cvt = [&](const float* s, _Float16* d, long long n) {
    int blocks = (int)((n + 255) / 256);
    if (blocks > 8192) blocks = 8192;
    cvt_f32_to_f16<<<blocks, 256, 0, stream>>>(s, d, n);
  };
  // Big parallel GEMMs: bias + f16 output, no K-split.
  auto gemm_big = [&](const _Float16* A, long long lda, const _Float16* W,
                      long long ldw, const float* bias, _Float16* Ch,
                      long long ldc, int M, int N, int K) {
    int groups = ((M >> 4) / 2) * ((N >> 4) / 2);
    dim3 grid((groups + 7) / 8, 1);    // 8 waves (256 threads) per block
    wmma_gemm_f16<2, 2, false><<<grid, 256, 0, stream>>>(
        A, lda, W, ldw, bias, Ch, ldc, M, N, K, nullptr);
  };
  // Scan GEMMs: K-split x4, raw f32 partials (reduced in fused epilogues).
  auto gemm_split = [&](const _Float16* A, long long lda, const _Float16* W,
                        long long ldw, float* Cpart, int M, int N, int K) {
    int groups = ((M >> 4) / 2) * ((N >> 4) / 2);
    dim3 grid((groups + 7) / 8, KSPLIT);
    wmma_gemm_f16<2, 2, true><<<grid, 256, 0, stream>>>(
        A, lda, W, ldw, nullptr, nullptr, 0, M, N, K, Cpart);
  };

  // One-time f16 conversions.
  cvt(x,     xh,    BS * Isz);
  cvt(W_in,  Winh,  (long long)Hsz * Isz);
  cvt(W_ih,  Wihh,  (long long)3 * Hsz * Hsz);
  cvt(W_hh,  Whhh,  (long long)3 * Hsz * Hsz);
  cvt(W_out, Wouth, (long long)Osz * Hsz);
  zero_state<<<(Bsz * Hsz + 255) / 256, 256, 0, stream>>>(hbuf, h16, Bsz * Hsz);

  // Parallel precompute: inp = x @ W_in^T + b_in ;  gx = inp @ W_ih^T + b_ih
  gemm_big(xh,   Isz, Winh, Isz, b_in, inph, Hsz,     (int)BS, Hsz,     Isz);
  gemm_big(inph, Hsz, Wihh, Hsz, b_ih, gxh,  3 * Hsz, (int)BS, 3 * Hsz, Hsz);

  // Sequential scan. W_hh/W_out (7 MB f16) stay L2-resident across all steps.
  // K-split x4 on both per-step GEMMs for 4x wave parallelism + short chains;
  // reductions fused into gru_gate_ln / out_reduce epilogues.
  for (int t = 0; t < Ssz; ++t) {
    gemm_split(h16, Hsz, Whhh, Hsz, ghp, Bsz, 3 * Hsz, Hsz);
    gru_gate_ln<<<Bsz, 256, 0, stream>>>(ghp, b_hh,
                                         gxh + (long long)t * 3 * Hsz,
                                         (long long)Ssz * 3 * Hsz,
                                         tau, gamma, beta, hbuf, h16);
    gemm_split(h16, Hsz, Wouth, Hsz, outp, Bsz, Osz, Hsz);
    out_reduce<<<(Bsz * Osz + 255) / 256, 256, 0, stream>>>(
        outp, b_out, out + (long long)t * Osz);
  }

  // h_final appended after outs.
  copy_f32<<<(Bsz * Hsz + 255) / 256, 256, 0, stream>>>(
      hbuf, out + (long long)Bsz * Ssz * Osz, Bsz * Hsz);
}
